// ProtoNetManifold_36481452212365
// MI455X (gfx1250) — compile-verified, matrix operand order checked by reading, not run on hardware
//
#include <hip/hip_runtime.h>

#define N_WAY   2
#define K_SHOT  1
#define N_PTS   2048
#define FEAT    320
#define N_Q     2
#define KNN     20
#define C_CLS   3
#define SIZE_S  2
#define N_TOTAL 8192
#define ALPHA_C 0.99f
#define SCALER  10.0f
#define CG_ITERS 5

typedef __bf16 bf16;
typedef bf16  v16bf __attribute__((ext_vector_type(16)));
typedef float v8f   __attribute__((ext_vector_type(8)));

// ---------------- helpers ----------------

__device__ inline unsigned short f32_to_bf16(float f) {
    unsigned int u = __float_as_uint(f);
    unsigned int r = u + 0x7FFFu + ((u >> 16) & 1u);
    return (unsigned short)(r >> 16);
}

__device__ inline float blockReduceSum(float v, float* sbuf) {
    int t = threadIdx.x;
    sbuf[t] = v;
    __syncthreads();
    for (int s = blockDim.x >> 1; s > 0; s >>= 1) {
        if (t < s) sbuf[t] += sbuf[t + s];
        __syncthreads();
    }
    float r = sbuf[0];
    __syncthreads();
    return r;
}

// ---------------- generic zero ----------------

__global__ void k_zero(float* p, int n) {
    int i = blockIdx.x * blockDim.x + threadIdx.x;
    if (i < n) p[i] = 0.f;
}

// ---------------- prototypes ----------------
// grid = FEAT blocks, 256 threads. protos: (C_CLS, FEAT): [0]=bg, [1..2]=fg per way.
__global__ void k_protos(const float* __restrict__ sfeat, const int* __restrict__ sy,
                         float* __restrict__ protos) {
    __shared__ float sbuf[256];
    int f = blockIdx.x;
    int t = threadIdx.x;
    float bgacc = 0.f;
    for (int w = 0; w < N_WAY; ++w) {
        float fs = 0.f, bs = 0.f, fc = 0.f, bc = 0.f;
        for (int p = t; p < N_PTS; p += 256) {
            float x = sfeat[((size_t)w * FEAT + f) * N_PTS + p];
            float y = (float)sy[w * N_PTS + p];
            fs += x * y;
            bs += x * (1.f - y);
            fc += y;
            bc += 1.f - y;
        }
        fs = blockReduceSum(fs, sbuf);
        bs = blockReduceSum(bs, sbuf);
        fc = blockReduceSum(fc, sbuf);
        bc = blockReduceSum(bc, sbuf);
        float fgf = fs / (fc + 1e-5f);   // K_SHOT==1 -> fg_protos row
        float bgf = bs / (bc + 1e-5f);
        bgacc += bgf;
        if (t == 0) protos[(1 + w) * FEAT + f] = fgf;
    }
    if (t == 0) protos[0 * FEAT + f] = bgacc / (float)(N_WAY * K_SHOT);
}

// grid = C_CLS blocks
__global__ void k_pn(const float* __restrict__ protos, float* __restrict__ pn) {
    __shared__ float sbuf[256];
    int c = blockIdx.x, t = threadIdx.x;
    float s = 0.f;
    for (int f = t; f < FEAT; f += 256) { float v = protos[c * FEAT + f]; s += v * v; }
    s = blockReduceSum(s, sbuf);
    float inv = 1.f / fmaxf(sqrtf(s), 1e-8f);
    for (int f = t; f < FEAT; f += 256) pn[c * FEAT + f] = protos[c * FEAT + f] * inv;
}

// query_pred0[q][c][p]
__global__ void k_pred0(const float* __restrict__ qfeat, const float* __restrict__ pn,
                        float* __restrict__ pred0) {
    int gid = blockIdx.x * blockDim.x + threadIdx.x;
    if (gid >= N_Q * N_PTS) return;
    int q = gid / N_PTS, p = gid % N_PTS;
    float nn = 0.f, d0 = 0.f, d1 = 0.f, d2 = 0.f;
    for (int f = 0; f < FEAT; ++f) {
        float x = qfeat[((size_t)q * FEAT + f) * N_PTS + p];
        nn += x * x;
        d0 += x * pn[f];
        d1 += x * pn[FEAT + f];
        d2 += x * pn[2 * FEAT + f];
    }
    float inv = SCALER / fmaxf(sqrtf(nn), 1e-8f);
    pred0[((size_t)q * C_CLS + 0) * N_PTS + p] = d0 * inv;
    pred0[((size_t)q * C_CLS + 1) * N_PTS + p] = d1 * inv;
    pred0[((size_t)q * C_CLS + 2) * N_PTS + p] = d2 * inv;
}

// row-normalize embeddings into bf16 matrix Xn (N_TOTAL x FEAT, row-major)
__global__ void k_xn(const float* __restrict__ sfeat, const float* __restrict__ qfeat,
                     unsigned short* __restrict__ xbf) {
    int r = blockIdx.x * blockDim.x + threadIdx.x;
    if (r >= N_TOTAL) return;
    const float* base;
    int p;
    if (r < SIZE_S * N_PTS) {
        int s = r / N_PTS; p = r % N_PTS;
        base = sfeat + (size_t)s * FEAT * N_PTS;
    } else {
        int q = (r - SIZE_S * N_PTS) / N_PTS; p = (r - SIZE_S * N_PTS) % N_PTS;
        base = qfeat + (size_t)q * FEAT * N_PTS;
    }
    float nn = 0.f;
    for (int f = 0; f < FEAT; ++f) { float x = base[(size_t)f * N_PTS + p]; nn += x * x; }
    float inv = 1.f / fmaxf(sqrtf(nn), 1e-12f);
    for (int f = 0; f < FEAT; ++f)
        xbf[(size_t)r * FEAT + f] = f32_to_bf16(base[(size_t)f * N_PTS + p] * inv);
}

// ---------------- fused sims (WMMA bf16) + top-k ----------------

__device__ inline v16bf load_frag(const unsigned short* __restrict__ xbf, int row, int kbase) {
    // 16-bit A layout: lanes 0-15 rows M=0..15 carry K octets [kbase..kbase+7] in v0-3
    // and [kbase+16..kbase+23] in v4-7; lanes 16-31 carry the +8 octets (kbase pre-offset).
    union { v16bf v; uint4 q[2]; } u;
    const unsigned short* p = xbf + (size_t)row * FEAT + kbase;
    u.q[0] = *reinterpret_cast<const uint4*>(p);
    u.q[1] = *reinterpret_cast<const uint4*>(p + 16);
    return u.v;
}

__global__ void __launch_bounds__(32) k_sims_topk(const unsigned short* __restrict__ xbf,
                                                  float* __restrict__ kvals,
                                                  int* __restrict__ kidx) {
    __shared__ float tile[16][16];
    __shared__ float tv[16][KNN];
    __shared__ int   ti[16][KNN];
    int lane = threadIdx.x;
    int r0 = blockIdx.x * 16;
    int lrow  = lane & 15;
    int mrow  = r0 + lrow;
    int khalf = (lane >> 4) << 3;   // 0 or 8: K-octet split across lane halves

    v16bf a[10];
#pragma unroll
    for (int kk = 0; kk < 10; ++kk) a[kk] = load_frag(xbf, mrow, kk * 32 + khalf);

    if (lane < 16) {
        for (int k = 0; k < KNN; ++k) { tv[lane][k] = -1e30f; ti[lane][k] = 0; }
    }
    __syncthreads();

    // Software pipeline, depth 2 with cross-tile prefetch: the B fragment fed
    // to WMMA kk was issued at step kk-2 (>=4 b128 loads always in flight),
    // and steps kk=8,9 prefetch fragments 0,1 of the NEXT column tile so those
    // loads overlap the tile-store / barrier / top-k tail of this iteration.
    v16bf p0 = load_frag(xbf, lrow, khalf);        // ct=0, frag 0
    v16bf p1 = load_frag(xbf, lrow, 32 + khalf);   // ct=0, frag 1

    for (int ct = 0; ct < N_TOTAL / 16; ++ct) {
        int crow = ct * 16 + lrow;
        int nrow = (ct + 1 < N_TOTAL / 16) ? crow + 16 : crow;  // clamp at last tile
        v8f acc;
#pragma unroll
        for (int r = 0; r < 8; ++r) acc[r] = 0.f;

        v16bf b0 = p0, b1 = p1, bn;
#pragma unroll
        for (int kk = 0; kk < 10; ++kk) {
            int pf = kk + 2;
            if (pf < 10) bn = load_frag(xbf, crow, pf * 32 + khalf);
            else         bn = load_frag(xbf, nrow, (pf - 10) * 32 + khalf);
            acc = __builtin_amdgcn_wmma_f32_16x16x32_bf16(
                false, a[kk], false, b0, (short)0, acc, false, false);
            b0 = b1; b1 = bn;
        }
        p0 = b0; p1 = b1;   // fragments 0,1 of next tile, already in flight

        int mbase = (lane < 16) ? 0 : 8;
        int n = lane & 15;
#pragma unroll
        for (int r = 0; r < 8; ++r) tile[mbase + r][n] = acc[r];
        __syncthreads();
        if (lane < 16) {
            int m = lane, grow = r0 + m;
            for (int n2 = 0; n2 < 16; ++n2) {
                int col = ct * 16 + n2;
                if (col == grow) continue;            // self excluded (== vals[:,0])
                float v = tile[m][n2];
                if (v > tv[m][KNN - 1]) {
                    int pos = KNN - 1;
                    while (pos > 0 && tv[m][pos - 1] < v) {
                        tv[m][pos] = tv[m][pos - 1];
                        ti[m][pos] = ti[m][pos - 1];
                        --pos;
                    }
                    tv[m][pos] = v; ti[m][pos] = col;
                }
            }
        }
        __syncthreads();
    }
    if (lane < 16) {
        int grow = r0 + lane;
        for (int k = 0; k < KNN; ++k) {
            kvals[grow * KNN + k] = tv[lane][k];
            kidx[grow * KNN + k]  = ti[lane][k];
        }
    }
}

// ---------------- graph construction + CG ----------------

__global__ void k_counts(const int* __restrict__ sy, int* counts) {
    int i = blockIdx.x * blockDim.x + threadIdx.x;
    if (i < SIZE_S * N_PTS) atomicAdd(&counts[sy[i]], 1);
}

__global__ void k_rsinit(const int* __restrict__ counts, float* rs, float* rs_new) {
    int c = threadIdx.x;
    if (c < C_CLS) {
        float v = counts[c] > 0 ? 1.f / (float)counts[c] : 0.f;  // sum(Y^2) per class
        rs[c] = v; rs_new[c] = v;
    }
}

__global__ void k_rpfill(const int* __restrict__ sy, const int* __restrict__ counts,
                         float* __restrict__ R, float* __restrict__ P) {
    int e = blockIdx.x * blockDim.x + threadIdx.x;
    if (e >= N_TOTAL * C_CLS) return;
    int i = e / C_CLS, c = e % C_CLS;
    float v = 0.f;
    if (i < SIZE_S * N_PTS) {
        if (sy[i] == c) v = 1.f / fmaxf((float)counts[c], 1.f);
    }
    R[e] = v; P[e] = v;
}

__global__ void k_deg(const float* __restrict__ kv, const int* __restrict__ ki,
                      float* __restrict__ deg) {
    int i = blockIdx.x * blockDim.x + threadIdx.x;
    if (i >= N_TOTAL) return;
    float s = 0.f;
    for (int k = 0; k < KNN; ++k) {
        float v = kv[i * KNN + k];
        float w = v * v * v;
        s += w;
        atomicAdd(&deg[ki[i * KNN + k]], w);
    }
    atomicAdd(&deg[i], s);
}

__global__ void k_dn(const float* __restrict__ deg, float* __restrict__ dn) {
    int i = blockIdx.x * blockDim.x + threadIdx.x;
    if (i >= N_TOTAL) return;
    float d = deg[i];
    if (d == 0.f) d = 1.f;
    dn[i] = 1.f / sqrtf(d);
}

// CG step 1: V2 = dn*P ; WV = 0 ; rs <- rs_new (carry) ; pAp = 0
__global__ void k_cg1(const float* __restrict__ dn, const float* __restrict__ P,
                      float* __restrict__ V2, float* __restrict__ WV,
                      float* rs, const float* rs_new, float* pAp) {
    int e = blockIdx.x * blockDim.x + threadIdx.x;
    if (e < C_CLS) { rs[e] = rs_new[e]; pAp[e] = 0.f; }
    if (e >= N_TOTAL * C_CLS) return;
    V2[e] = dn[e / C_CLS] * P[e];
    WV[e] = 0.f;
}

// CG step 2: symmetric edge scatter WV = W_mv(V2)
__global__ void k_cg2(const float* __restrict__ kv, const int* __restrict__ ki,
                      const float* __restrict__ V2, float* __restrict__ WV) {
    int e = blockIdx.x * blockDim.x + threadIdx.x;
    if (e >= N_TOTAL * KNN) return;
    int i = e / KNN;
    int j = ki[e];
    float v = kv[e];
    float w = v * v * v;
    for (int c = 0; c < C_CLS; ++c) {
        atomicAdd(&WV[i * C_CLS + c], w * V2[j * C_CLS + c]);
        atomicAdd(&WV[j * C_CLS + c], w * V2[i * C_CLS + c]);
    }
}

// CG step 3: AP = P - alpha*dn*WV ; pAp += P.AP ; rs_new = 0
__global__ void k_cg3(const float* __restrict__ P, const float* __restrict__ dn,
                      const float* __restrict__ WV, float* __restrict__ AP,
                      float* pAp, float* rs_new) {
    int e = blockIdx.x * blockDim.x + threadIdx.x;
    if (e < C_CLS) rs_new[e] = 0.f;
    if (e >= N_TOTAL * C_CLS) return;
    float ap = P[e] - ALPHA_C * dn[e / C_CLS] * WV[e];
    AP[e] = ap;
    atomicAdd(&pAp[e % C_CLS], P[e] * ap);
}

// CG step 4: alpha = rs/(pAp+eps) ; X += alpha*P ; R -= alpha*AP ; rs_new += R^2
__global__ void k_cg4(const float* rs, const float* pAp,
                      const float* __restrict__ P, const float* __restrict__ AP,
                      float* __restrict__ Xc, float* __restrict__ R, float* rs_new) {
    int e = blockIdx.x * blockDim.x + threadIdx.x;
    if (e >= N_TOTAL * C_CLS) return;
    int c = e % C_CLS;
    float alpha = rs[c] / (pAp[c] + 1e-30f);
    Xc[e] += alpha * P[e];
    float r = R[e] - alpha * AP[e];
    R[e] = r;
    atomicAdd(&rs_new[c], r * r);
}

// CG step 5: beta = rs_new/(rs+eps) ; P = R + beta*P
__global__ void k_cg5(const float* rs, const float* rs_new,
                      const float* __restrict__ R, float* __restrict__ P) {
    int e = blockIdx.x * blockDim.x + threadIdx.x;
    if (e >= N_TOTAL * C_CLS) return;
    int c = e % C_CLS;
    float beta = rs_new[c] / (rs[c] + 1e-30f);
    P[e] = R[e] + beta * P[e];
}

// ---------------- predictions + losses ----------------

__global__ void k_predloss(const float* __restrict__ Xc, const int* __restrict__ sy,
                           const int* __restrict__ qy, float* __restrict__ pred,
                           float* loss_m) {
    int i = blockIdx.x * blockDim.x + threadIdx.x;
    if (i >= N_TOTAL) return;
    float z0 = fmaxf(Xc[i * 3 + 0], 0.f);
    float z1 = fmaxf(Xc[i * 3 + 1], 0.f);
    float z2 = fmaxf(Xc[i * 3 + 2], 0.f);
    float inv = 1.f / fmaxf(z0 + z1 + z2, 1e-12f);
    float p0 = z0 * inv, p1 = z1 * inv, p2 = z2 * inv;
    pred[i * 3 + 0] = p0; pred[i * 3 + 1] = p1; pred[i * 3 + 2] = p2;
    int label = (i < SIZE_S * N_PTS) ? sy[i] : qy[i - SIZE_S * N_PTS];
    float m = fmaxf(p0, fmaxf(p1, p2));
    float lse = m + logf(expf(p0 - m) + expf(p1 - m) + expf(p2 - m));
    float lp = ((label == 0) ? p0 : (label == 1) ? p1 : p2) - lse;
    atomicAdd(loss_m, -lp * (1.f / (float)N_TOTAL));
}

__global__ void k_lossproto(const float* __restrict__ pred0, const int* __restrict__ qy,
                            float* loss_p) {
    int gid = blockIdx.x * blockDim.x + threadIdx.x;
    if (gid >= N_Q * N_PTS) return;
    int q = gid / N_PTS, p = gid % N_PTS;
    float x0 = pred0[((size_t)q * C_CLS + 0) * N_PTS + p];
    float x1 = pred0[((size_t)q * C_CLS + 1) * N_PTS + p];
    float x2 = pred0[((size_t)q * C_CLS + 2) * N_PTS + p];
    int label = qy[gid];
    float m = fmaxf(x0, fmaxf(x1, x2));
    float lse = m + logf(expf(x0 - m) + expf(x1 - m) + expf(x2 - m));
    float lp = ((label == 0) ? x0 : (label == 1) ? x1 : x2) - lse;
    atomicAdd(loss_p, -lp * (1.f / (float)(N_Q * N_PTS)));
}

__global__ void k_out(const float* __restrict__ pred, const float* __restrict__ pred0,
                      const float* loss_m, const float* loss_p, float* __restrict__ out) {
    int gid = blockIdx.x * blockDim.x + threadIdx.x;
    if (gid < N_Q * C_CLS * N_PTS) {
        int q = gid / (C_CLS * N_PTS);
        int c = (gid / N_PTS) % C_CLS;
        int p = gid % N_PTS;
        int row = SIZE_S * N_PTS + q * N_PTS + p;
        out[gid] = 0.5f * (pred[row * C_CLS + c] + pred0[gid]);
    } else if (gid == N_Q * C_CLS * N_PTS) {
        out[gid] = loss_m[0] + loss_p[0];
    }
}

// ---------------- launch ----------------

extern "C" void kernel_launch(void* const* d_in, const int* in_sizes, int n_in,
                              void* d_out, int out_size, void* d_ws, size_t ws_size,
                              hipStream_t stream) {
    const float* sfeat = (const float*)d_in[0];
    const float* qfeat = (const float*)d_in[1];
    const int*   sy    = (const int*)d_in[2];
    const int*   qy    = (const int*)d_in[3];
    float* out = (float*)d_out;
    char* ws = (char*)d_ws;

    // workspace layout (all 16B-aligned)
    constexpr size_t OFF_PROTOS = 0;
    constexpr size_t OFF_PN     = OFF_PROTOS + (size_t)C_CLS * FEAT * 4;
    constexpr size_t OFF_PRED0  = OFF_PN + (size_t)C_CLS * FEAT * 4;
    constexpr size_t OFF_XBF    = OFF_PRED0 + (size_t)N_Q * C_CLS * N_PTS * 4;
    constexpr size_t OFF_KVALS  = OFF_XBF + (size_t)N_TOTAL * FEAT * 2;
    constexpr size_t OFF_KIDX   = OFF_KVALS + (size_t)N_TOTAL * KNN * 4;
    constexpr size_t OFF_DEG    = OFF_KIDX + (size_t)N_TOTAL * KNN * 4;
    constexpr size_t OFF_DN     = OFF_DEG + (size_t)N_TOTAL * 4;
    constexpr size_t OFF_R      = OFF_DN + (size_t)N_TOTAL * 4;
    constexpr size_t OFF_P      = OFF_R + (size_t)N_TOTAL * C_CLS * 4;
    constexpr size_t OFF_XC     = OFF_P + (size_t)N_TOTAL * C_CLS * 4;
    constexpr size_t OFF_AP     = OFF_XC + (size_t)N_TOTAL * C_CLS * 4;
    constexpr size_t OFF_V2     = OFF_AP + (size_t)N_TOTAL * C_CLS * 4;
    constexpr size_t OFF_WV     = OFF_V2 + (size_t)N_TOTAL * C_CLS * 4;
    constexpr size_t OFF_PRED   = OFF_WV + (size_t)N_TOTAL * C_CLS * 4;
    constexpr size_t OFF_SCAL   = OFF_PRED + (size_t)N_TOTAL * C_CLS * 4;

    float* protos = (float*)(ws + OFF_PROTOS);
    float* pn     = (float*)(ws + OFF_PN);
    float* pred0  = (float*)(ws + OFF_PRED0);
    unsigned short* xbf = (unsigned short*)(ws + OFF_XBF);
    float* kvals  = (float*)(ws + OFF_KVALS);
    int*   kidx   = (int*)(ws + OFF_KIDX);
    float* deg    = (float*)(ws + OFF_DEG);
    float* dn     = (float*)(ws + OFF_DN);
    float* R      = (float*)(ws + OFF_R);
    float* P      = (float*)(ws + OFF_P);
    float* Xc     = (float*)(ws + OFF_XC);
    float* AP     = (float*)(ws + OFF_AP);
    float* V2     = (float*)(ws + OFF_V2);
    float* WV     = (float*)(ws + OFF_WV);
    float* pred   = (float*)(ws + OFF_PRED);
    float* scal   = (float*)(ws + OFF_SCAL);
    int*   counts = (int*)(scal + 0);   // [0..3]
    float* rs     = scal + 4;           // [4..6]
    float* rs_new = scal + 8;           // [8..10]
    float* pAp    = scal + 12;          // [12..14]
    float* loss_m = scal + 16;
    float* loss_p = scal + 17;

    const int B = 256;
    const int NE = N_TOTAL * C_CLS;   // 24576

    // init
    k_zero<<<(N_TOTAL + B - 1) / B, B, 0, stream>>>(deg, N_TOTAL);
    k_zero<<<(NE + B - 1) / B, B, 0, stream>>>(Xc, NE);
    k_zero<<<1, 32, 0, stream>>>(scal, 32);

    // prototypes + proto scores
    k_protos<<<FEAT, B, 0, stream>>>(sfeat, sy, protos);
    k_pn<<<C_CLS, B, 0, stream>>>(protos, pn);
    k_pred0<<<(N_Q * N_PTS + B - 1) / B, B, 0, stream>>>(qfeat, pn, pred0);

    // normalized bf16 embedding matrix
    k_xn<<<(N_TOTAL + B - 1) / B, B, 0, stream>>>(sfeat, qfeat, xbf);

    // fused sims (WMMA) + top-k graph
    k_sims_topk<<<N_TOTAL / 16, 32, 0, stream>>>(xbf, kvals, kidx);

    // label setup
    k_counts<<<(SIZE_S * N_PTS + B - 1) / B, B, 0, stream>>>(sy, counts);
    k_rsinit<<<1, 32, 0, stream>>>(counts, rs, rs_new);
    k_rpfill<<<(NE + B - 1) / B, B, 0, stream>>>(sy, counts, R, P);

    // degrees
    k_deg<<<(N_TOTAL + B - 1) / B, B, 0, stream>>>(kvals, kidx, deg);
    k_dn<<<(N_TOTAL + B - 1) / B, B, 0, stream>>>(deg, dn);

    // conjugate gradient (5 iterations, scalars carried in ws)
    for (int it = 0; it < CG_ITERS; ++it) {
        k_cg1<<<(NE + B - 1) / B, B, 0, stream>>>(dn, P, V2, WV, rs, rs_new, pAp);
        k_cg2<<<(N_TOTAL * KNN + B - 1) / B, B, 0, stream>>>(kvals, kidx, V2, WV);
        k_cg3<<<(NE + B - 1) / B, B, 0, stream>>>(P, dn, WV, AP, pAp, rs_new);
        k_cg4<<<(NE + B - 1) / B, B, 0, stream>>>(rs, pAp, P, AP, Xc, R, rs_new);
        k_cg5<<<(NE + B - 1) / B, B, 0, stream>>>(rs, rs_new, R, P);
    }

    // predictions, losses, output
    k_predloss<<<(N_TOTAL + B - 1) / B, B, 0, stream>>>(Xc, sy, qy, pred, loss_m);
    k_lossproto<<<(N_Q * N_PTS + B - 1) / B, B, 0, stream>>>(pred0, qy, loss_p);
    k_out<<<(N_Q * C_CLS * N_PTS + 1 + B - 1) / B, B, 0, stream>>>(pred, pred0, loss_m, loss_p, out);
}
